// TransformerBlock_14078902796833
// MI455X (gfx1250) — compile-verified, matrix operand order checked by reading, not run on hardware
//
#include <hip/hip_runtime.h>

typedef __attribute__((ext_vector_type(16))) __bf16 v16bf;
typedef __attribute__((ext_vector_type(8)))  float  v8f;

static constexpr int kN = 4, kL = 1024, kD = 1024, kH = 16;

// ---------------------------------------------------------------------------
// WMMA fragment loaders.
// 16-bit A-matrix 16x32 layout (ISA 7.12.2): lanes 0-15 hold M=lane with
// K={0..7,16..23}; lanes 16-31 hold M=lane-16 with K={8..15,24..31}.
// Elements 0..7 of the v16bf fragment are the first K-octet, 8..15 the second.
// B (32x16) is loaded with the mirrored layout (lane = N column = row of W).
// ---------------------------------------------------------------------------
union FragCast { uint4 u[2]; v16bf v; };

__device__ __forceinline__ v16bf load_frag_bf16(const __bf16* base, int row,
                                                int ld, int k0, int ksel) {
  const __bf16* q = base + (size_t)row * ld + k0 + 8 * ksel;
  FragCast t;
  t.u[0] = *(const uint4*)(q);
  t.u[1] = *(const uint4*)(q + 16);
  return t.v;
}

__device__ __forceinline__ v16bf load_frag_f32(const float* base, int row,
                                               int ld, int k0, int ksel) {
  const float* q = base + (size_t)row * ld + k0 + 8 * ksel;
  float4 f0 = ((const float4*)q)[0];
  float4 f1 = ((const float4*)q)[1];
  float4 f2 = ((const float4*)(q + 16))[0];
  float4 f3 = ((const float4*)(q + 16))[1];
  v16bf v;
  v[0]  = (__bf16)f0.x; v[1]  = (__bf16)f0.y; v[2]  = (__bf16)f0.z; v[3]  = (__bf16)f0.w;
  v[4]  = (__bf16)f1.x; v[5]  = (__bf16)f1.y; v[6]  = (__bf16)f1.z; v[7]  = (__bf16)f1.w;
  v[8]  = (__bf16)f2.x; v[9]  = (__bf16)f2.y; v[10] = (__bf16)f2.z; v[11] = (__bf16)f2.w;
  v[12] = (__bf16)f3.x; v[13] = (__bf16)f3.y; v[14] = (__bf16)f3.z; v[15] = (__bf16)f3.w;
  return v;
}

enum { GF_CBF16 = 1, GF_RELU = 2, GF_TRANSC = 4 };

__device__ __forceinline__ void store_c_elem(void* C, long long idx, float v,
                                             const float* bias, int col, int flags) {
  if (bias) v += bias[col];
  if (flags & GF_RELU) v = fmaxf(v, 0.f);
  if (flags & GF_CBF16) ((__bf16*)C)[idx] = (__bf16)v;
  else                  ((float*)C)[idx]  = v;
}

// ---------------------------------------------------------------------------
// Generic batched GEMM:  C[b] = A[b] (MxK, row-major) * B[b]^T (B is NxK
// row-major, i.e. torch Linear weight layout).  Two batch dims (b0,b1) with
// independent strides so (n,h)-batched attention GEMMs map directly.
// Block = 128 threads = 4 waves (2x2), wave tile 32x32, block tile 64x64.
// Used for the N=64 / batched attention GEMMs.
// ---------------------------------------------------------------------------
struct GemmParams {
  const void* A; const void* B; void* C; const float* bias;
  int M, N, K, lda, ldb, ldc;
  long long sA0, sA1, sB0, sB1, sC0, sC1;
  int nb1, flags;
};

template <bool A_F32>
__global__ __launch_bounds__(128) void gemm_wmma_kernel(GemmParams p) {
  const int lane = threadIdx.x & 31;
  const int wave = threadIdx.x >> 5;
  const int b1 = (int)(blockIdx.z) % p.nb1;
  const int b0 = (int)(blockIdx.z) / p.nb1;

  const __bf16* B = (const __bf16*)p.B + b0 * p.sB0 + b1 * p.sB1;
  const float*  Af = (const float*)p.A + b0 * p.sA0 + b1 * p.sA1;
  const __bf16* Ab = (const __bf16*)p.A + b0 * p.sA0 + b1 * p.sA1;

  const int m0 = blockIdx.y * 64 + (wave >> 1) * 32;
  const int n0 = blockIdx.x * 64 + (wave & 1) * 32;
  const int r15 = lane & 15;
  const int ksel = lane >> 4;

  v8f acc[2][2];
#pragma unroll
  for (int mi = 0; mi < 2; ++mi)
#pragma unroll
    for (int ni = 0; ni < 2; ++ni)
#pragma unroll
      for (int i = 0; i < 8; ++i) acc[mi][ni][i] = 0.f;

#pragma unroll 2
  for (int k0 = 0; k0 < p.K; k0 += 32) {
    // L2 prefetch two K-steps ahead (global_prefetch_b8 path).
    if (k0 + 64 < p.K) {
      if (A_F32) __builtin_prefetch(Af + (size_t)(m0 + r15) * p.lda + k0 + 64, 0, 1);
      else       __builtin_prefetch(Ab + (size_t)(m0 + r15) * p.lda + k0 + 64, 0, 1);
      __builtin_prefetch(B + (size_t)(n0 + r15) * p.ldb + k0 + 64, 0, 1);
    }
    v16bf af[2], bf[2];
    if (A_F32) {
      af[0] = load_frag_f32(Af, m0 + r15,      p.lda, k0, ksel);
      af[1] = load_frag_f32(Af, m0 + 16 + r15, p.lda, k0, ksel);
    } else {
      af[0] = load_frag_bf16(Ab, m0 + r15,      p.lda, k0, ksel);
      af[1] = load_frag_bf16(Ab, m0 + 16 + r15, p.lda, k0, ksel);
    }
    bf[0] = load_frag_bf16(B, n0 + r15,      p.ldb, k0, ksel);
    bf[1] = load_frag_bf16(B, n0 + 16 + r15, p.ldb, k0, ksel);
#pragma unroll
    for (int mi = 0; mi < 2; ++mi)
#pragma unroll
      for (int ni = 0; ni < 2; ++ni)
        acc[mi][ni] = __builtin_amdgcn_wmma_f32_16x16x32_bf16(
            false, af[mi], false, bf[ni], (short)0, acc[mi][ni], false, false);
  }

  // C layout (32-bit 16x16): lanes 0-15 rows 0-7 in VGPR r, lanes 16-31 rows 8-15.
  const int cn = lane & 15;
  const int cm = (lane >> 4) * 8;
  const long long coff = b0 * p.sC0 + b1 * p.sC1;
  char* Cb = (char*)p.C + coff * ((p.flags & GF_CBF16) ? 2 : 4);
#pragma unroll
  for (int mi = 0; mi < 2; ++mi)
#pragma unroll
    for (int ni = 0; ni < 2; ++ni)
#pragma unroll
      for (int r = 0; r < 8; ++r) {
        const int row = m0 + mi * 16 + cm + r;
        const int col = n0 + ni * 16 + cn;
        const long long idx =
            (p.flags & GF_TRANSC) ? ((long long)col * p.ldc + row)
                                  : ((long long)row * p.ldc + col);
        store_c_elem(Cb, idx, acc[mi][ni][r], p.bias, col, p.flags);
      }
}

// ---------------------------------------------------------------------------
// High-intensity GEMM for the large unbatched matmuls (GRU gates, w_out, fc):
// M,N multiples of 128, K multiple of 64.  Block = 128 threads = 4 waves
// (2x2), wave tile 64x64 -> 16 WMMAs per 16 b128 fragment loads per K-step
// (1 VMEM per WMMA), K-unrolled x2 so loads of step i+1 overlap the WMMA
// chain of step i.
// ---------------------------------------------------------------------------
struct GemmBigParams {
  const __bf16* A; const __bf16* B; void* C; const float* bias;
  int K, lda, ldb, ldc, flags;
};

__global__ __launch_bounds__(128) void gemm_wmma_big_kernel(GemmBigParams p) {
  const int lane = threadIdx.x & 31;
  const int wave = threadIdx.x >> 5;
  const int m0 = blockIdx.y * 128 + (wave >> 1) * 64;
  const int n0 = blockIdx.x * 128 + (wave & 1) * 64;
  const int r15 = lane & 15;
  const int ksel = lane >> 4;

  v8f acc[4][4];
#pragma unroll
  for (int mi = 0; mi < 4; ++mi)
#pragma unroll
    for (int ni = 0; ni < 4; ++ni)
#pragma unroll
      for (int i = 0; i < 8; ++i) acc[mi][ni][i] = 0.f;

#pragma unroll 2
  for (int k0 = 0; k0 < p.K; k0 += 32) {
    if (k0 + 64 < p.K) {
      __builtin_prefetch(p.A + (size_t)(m0 + r15) * p.lda + k0 + 64, 0, 1);
      __builtin_prefetch(p.B + (size_t)(n0 + r15) * p.ldb + k0 + 64, 0, 1);
    }
    v16bf af[4], bf[4];
#pragma unroll
    for (int i = 0; i < 4; ++i)
      af[i] = load_frag_bf16(p.A, m0 + 16 * i + r15, p.lda, k0, ksel);
#pragma unroll
    for (int i = 0; i < 4; ++i)
      bf[i] = load_frag_bf16(p.B, n0 + 16 * i + r15, p.ldb, k0, ksel);
#pragma unroll
    for (int mi = 0; mi < 4; ++mi)
#pragma unroll
      for (int ni = 0; ni < 4; ++ni)
        acc[mi][ni] = __builtin_amdgcn_wmma_f32_16x16x32_bf16(
            false, af[mi], false, bf[ni], (short)0, acc[mi][ni], false, false);
  }

  const int cn = lane & 15;
  const int cm = (lane >> 4) * 8;
#pragma unroll
  for (int mi = 0; mi < 4; ++mi)
#pragma unroll
    for (int ni = 0; ni < 4; ++ni)
#pragma unroll
      for (int r = 0; r < 8; ++r) {
        const int row = m0 + mi * 16 + cm + r;
        const int col = n0 + ni * 16 + cn;
        store_c_elem(p.C, (long long)row * p.ldc + col, acc[mi][ni][r],
                     p.bias, col, p.flags);
      }
}

// ---------------------------------------------------------------------------
// Block reductions (256 threads = 8 waves, wave32 shuffles + LDS)
// ---------------------------------------------------------------------------
__device__ __forceinline__ float block_sum(float v, float* sbuf) {
#pragma unroll
  for (int o = 16; o > 0; o >>= 1) v += __shfl_down(v, o, 32);
  if ((threadIdx.x & 31) == 0) sbuf[threadIdx.x >> 5] = v;
  __syncthreads();
  float r = (threadIdx.x < 8) ? sbuf[threadIdx.x] : 0.f;
  if (threadIdx.x < 32) {
#pragma unroll
    for (int o = 4; o > 0; o >>= 1) r += __shfl_down(r, o, 32);
    if (threadIdx.x == 0) sbuf[0] = r;
  }
  __syncthreads();
  r = sbuf[0];
  __syncthreads();
  return r;
}

__device__ __forceinline__ float block_max(float v, float* sbuf) {
#pragma unroll
  for (int o = 16; o > 0; o >>= 1) v = fmaxf(v, __shfl_down(v, o, 32));
  if ((threadIdx.x & 31) == 0) sbuf[threadIdx.x >> 5] = v;
  __syncthreads();
  float r = (threadIdx.x < 8) ? sbuf[threadIdx.x] : -3.0e38f;
  if (threadIdx.x < 32) {
#pragma unroll
    for (int o = 4; o > 0; o >>= 1) r = fmaxf(r, __shfl_down(r, o, 32));
    if (threadIdx.x == 0) sbuf[0] = r;
  }
  __syncthreads();
  r = sbuf[0];
  __syncthreads();
  return r;
}

// ---------------------------------------------------------------------------
// LayerNorm over last dim (D=1024): one block per row, f32 in -> bf16 out
// ---------------------------------------------------------------------------
__global__ __launch_bounds__(256) void ln_kernel(const float* __restrict__ x,
                                                 const float* __restrict__ s,
                                                 const float* __restrict__ b,
                                                 __bf16* __restrict__ out) {
  __shared__ float sbuf[8];
  const float* xr = x + (size_t)blockIdx.x * kD;
  float4 v = ((const float4*)xr)[threadIdx.x];
  float sum = v.x + v.y + v.z + v.w;
  float sq  = v.x * v.x + v.y * v.y + v.z * v.z + v.w * v.w;
  sum = block_sum(sum, sbuf);
  sq  = block_sum(sq, sbuf);
  const float mean = sum * (1.f / kD);
  const float var  = sq * (1.f / kD) - mean * mean;
  const float rstd = rsqrtf(var + 1e-5f);
  float4 sv = ((const float4*)s)[threadIdx.x];
  float4 bv = ((const float4*)b)[threadIdx.x];
  __bf16* o = out + (size_t)blockIdx.x * kD + threadIdx.x * 4;
  o[0] = (__bf16)((v.x - mean) * rstd * sv.x + bv.x);
  o[1] = (__bf16)((v.y - mean) * rstd * sv.y + bv.y);
  o[2] = (__bf16)((v.z - mean) * rstd * sv.z + bv.z);
  o[3] = (__bf16)((v.w - mean) * rstd * sv.w + bv.w);
}

// ---------------------------------------------------------------------------
// Masked, scaled softmax over k (L=1024), in place on the attn output region.
// energy scaled by 1/sqrt(D)=1/32 after masking with -1e20 (ref semantics).
// ---------------------------------------------------------------------------
__global__ __launch_bounds__(256) void softmax_kernel(float* __restrict__ e,
                                                      const int* __restrict__ mask) {
  __shared__ float sbuf[8];
  const int n = blockIdx.x >> 14;  // row = ((n*H + h)*L + q), H*L = 16384
  float* er = e + (size_t)blockIdx.x * kL;
  const int* mr = mask + n * kL;
  float4 v = ((const float4*)er)[threadIdx.x];
  int4 m = ((const int4*)mr)[threadIdx.x];
  const float kMasked = -1e20f * 0.03125f;
  float t0 = m.x ? v.x * 0.03125f : kMasked;
  float t1 = m.y ? v.y * 0.03125f : kMasked;
  float t2 = m.z ? v.z * 0.03125f : kMasked;
  float t3 = m.w ? v.w * 0.03125f : kMasked;
  float mx = block_max(fmaxf(fmaxf(t0, t1), fmaxf(t2, t3)), sbuf);
  float e0 = __expf(t0 - mx), e1 = __expf(t1 - mx);
  float e2 = __expf(t2 - mx), e3 = __expf(t3 - mx);
  float inv = 1.f / block_sum(e0 + e1 + e2 + e3, sbuf);
  ((float4*)er)[threadIdx.x] = make_float4(e0 * inv, e1 * inv, e2 * inv, e3 * inv);
}

// ---------------------------------------------------------------------------
// GRU gate elementwise halves. ywc: [M,3072] = y@[W0;W1;W2]^T,
// xuc: [M,2048] = x@[U0;U1]^T.
// ---------------------------------------------------------------------------
__device__ __forceinline__ float sigmoidf(float x) { return 1.f / (1.f + __expf(-x)); }

__global__ __launch_bounds__(256) void gru_pre_kernel(const float* __restrict__ ywc,
                                                      const float* __restrict__ xuc,
                                                      const float* __restrict__ x,
                                                      const float* __restrict__ bg,
                                                      __bf16* __restrict__ rx,
                                                      float* __restrict__ z) {
  const size_t total = (size_t)kN * kL * kD;
  const size_t stride = (size_t)gridDim.x * blockDim.x;
  for (size_t i = (size_t)blockIdx.x * blockDim.x + threadIdx.x; i < total; i += stride) {
    const size_t row = i >> 10;
    const int col = (int)(i & 1023);
    const float r = sigmoidf(ywc[row * 3072 + col] + xuc[row * 2048 + col]);
    const float zz = sigmoidf(ywc[row * 3072 + 1024 + col] +
                              xuc[row * 2048 + 1024 + col] - bg[col]);
    rx[i] = (__bf16)(r * x[i]);
    z[i] = zz;
  }
}

__global__ __launch_bounds__(256) void gru_comb_kernel(const float* __restrict__ ywc,
                                                       const float* __restrict__ rxu,
                                                       const float* __restrict__ z,
                                                       const float* __restrict__ x,
                                                       float* __restrict__ out,
                                                       __bf16* __restrict__ hb) {
  const size_t total = (size_t)kN * kL * kD;
  const size_t stride = (size_t)gridDim.x * blockDim.x;
  for (size_t i = (size_t)blockIdx.x * blockDim.x + threadIdx.x; i < total; i += stride) {
    const size_t row = i >> 10;
    const int col = (int)(i & 1023);
    const float hc = tanhf(ywc[row * 3072 + 2048 + col] + rxu[i]);
    const float zi = z[i];
    const float o = (1.f - zi) * x[i] + zi * hc;
    out[i] = o;
    if (hb) hb[i] = (__bf16)o;
  }
}

__global__ __launch_bounds__(256) void cvt_kernel(const float* __restrict__ src,
                                                  __bf16* __restrict__ dst, size_t n) {
  const size_t stride = (size_t)gridDim.x * blockDim.x;
  for (size_t i = (size_t)blockIdx.x * blockDim.x + threadIdx.x; i < n; i += stride)
    dst[i] = (__bf16)src[i];
}

// ---------------------------------------------------------------------------
// Host orchestration
// ---------------------------------------------------------------------------
static inline void launch_gemm(hipStream_t st, const void* A, bool af32,
                               const void* B, void* C, const float* bias,
                               int M, int N, int K, int lda, int ldb, int ldc,
                               long long sA0, long long sA1,
                               long long sB0, long long sB1,
                               long long sC0, long long sC1,
                               int nb0, int nb1, int flags) {
  GemmParams p{A, B, C, bias, M, N, K, lda, ldb, ldc,
               sA0, sA1, sB0, sB1, sC0, sC1, nb1, flags};
  dim3 grid(N / 64, M / 64, nb0 * nb1);
  if (af32) gemm_wmma_kernel<true><<<grid, 128, 0, st>>>(p);
  else      gemm_wmma_kernel<false><<<grid, 128, 0, st>>>(p);
}

static inline void launch_gemm_big(hipStream_t st, const __bf16* A, const __bf16* B,
                                   void* C, const float* bias,
                                   int M, int N, int K, int lda, int ldb, int ldc,
                                   int flags) {
  GemmBigParams p{A, B, C, bias, K, lda, ldb, ldc, flags};
  dim3 grid(N / 128, M / 128, 1);
  gemm_wmma_big_kernel<<<grid, 128, 0, st>>>(p);
}

extern "C" void kernel_launch(void* const* d_in, const int* in_sizes, int n_in,
                              void* d_out, int out_size, void* d_ws, size_t ws_size,
                              hipStream_t stream) {
  (void)in_sizes; (void)n_in; (void)out_size; (void)ws_size;
  const float* value  = (const float*)d_in[0];
  // d_in[1] ('key') is unused by the reference: k_ = LN(value).
  const float* query  = (const float*)d_in[2];
  const int*   mask   = (const int*)d_in[3];
  const float* wq     = (const float*)d_in[4];
  const float* wk     = (const float*)d_in[5];
  const float* wv     = (const float*)d_in[6];
  const float* w_out  = (const float*)d_in[7];
  const float* b_out  = (const float*)d_in[8];
  const float* fc_w   = (const float*)d_in[9];
  const float* fc_b   = (const float*)d_in[10];
  const float* ln1_s  = (const float*)d_in[11];
  const float* ln1_b  = (const float*)d_in[12];
  const float* lnkv_s = (const float*)d_in[13];
  const float* lnkv_b = (const float*)d_in[14];
  const float* ln2_s  = (const float*)d_in[15];
  const float* ln2_b  = (const float*)d_in[16];
  const float* g1_W   = (const float*)d_in[17];
  const float* g1_U   = (const float*)d_in[18];
  const float* g1_bg  = (const float*)d_in[19];
  const float* g2_W   = (const float*)d_in[20];
  const float* g2_U   = (const float*)d_in[21];
  const float* g2_bg  = (const float*)d_in[22];

  float* out  = (float*)d_out;                        // [N,L,D]
  float* attn = out + (size_t)kN * kL * kD;           // [N,H,L,L]

  // ---- workspace carve ----
  char* ws = (char*)d_ws;
  size_t off = 0;
  auto balloc = [&](size_t elems) {
    __bf16* r = (__bf16*)(ws + off);
    off += ((elems * 2 + 255) & ~(size_t)255);
    return r;
  };
  auto falloc = [&](size_t elems) {
    float* r = (float*)(ws + off);
    off += ((elems * 4 + 255) & ~(size_t)255);
    return r;
  };
  const size_t M4 = (size_t)kN * kL * kD;       // 4,194,304
  __bf16* wq_b   = balloc(64 * 64);
  __bf16* wk_b   = balloc(64 * 64);
  __bf16* wv_b   = balloc(64 * 64);
  __bf16* wout_b = balloc((size_t)kD * kD);
  __bf16* fcw_b  = balloc((size_t)kD * kD);
  __bf16* g1W_b  = balloc(3 * (size_t)kD * kD);
  __bf16* g1U_b  = balloc(3 * (size_t)kD * kD);
  __bf16* g2W_b  = balloc(3 * (size_t)kD * kD);
  __bf16* g2U_b  = balloc(3 * (size_t)kD * kD);
  __bf16* qx_b   = balloc(M4);   // raw query in bf16 (GRU1 x-side)
  __bf16* qn_b   = balloc(M4);   // LN(query)
  __bf16* vn_b   = balloc(M4);   // LN(value)
  __bf16* qh_b   = balloc(M4);
  __bf16* kh_b   = balloc(M4);
  __bf16* vhT_b  = balloc(M4);   // [n][h][d][q]
  __bf16* o_b    = balloc(M4);   // attn @ v  -> [n,l,h,d]
  __bf16* y1_b   = balloc(M4);   // o @ w_out^T + b_out
  __bf16* rx_b   = balloc(M4);
  __bf16* h_b    = balloc(M4);
  __bf16* hn_b   = balloc(M4);
  __bf16* f_b    = balloc(M4);
  float*  ywc    = falloc((size_t)kN * kL * 3 * kD);
  float*  xuc    = falloc((size_t)kN * kL * 2 * kD);
  float*  zbuf   = falloc(M4);
  float*  rxu    = falloc(M4);
  float*  hbuf   = falloc(M4);

  // ---- weight / activation conversions to bf16 ----
  cvt_kernel<<<64,   256, 0, stream>>>(wq, wq_b, 64 * 64);
  cvt_kernel<<<64,   256, 0, stream>>>(wk, wk_b, 64 * 64);
  cvt_kernel<<<64,   256, 0, stream>>>(wv, wv_b, 64 * 64);
  cvt_kernel<<<1024, 256, 0, stream>>>(w_out, wout_b, (size_t)kD * kD);
  cvt_kernel<<<1024, 256, 0, stream>>>(fc_w, fcw_b, (size_t)kD * kD);
  cvt_kernel<<<2048, 256, 0, stream>>>(g1_W, g1W_b, 3 * (size_t)kD * kD);
  cvt_kernel<<<2048, 256, 0, stream>>>(g1_U, g1U_b, 3 * (size_t)kD * kD);
  cvt_kernel<<<2048, 256, 0, stream>>>(g2_W, g2W_b, 3 * (size_t)kD * kD);
  cvt_kernel<<<2048, 256, 0, stream>>>(g2_U, g2U_b, 3 * (size_t)kD * kD);
  cvt_kernel<<<2048, 256, 0, stream>>>(query, qx_b, M4);

  // ---- pre-LN ----
  ln_kernel<<<kN * kL, 256, 0, stream>>>(query, ln1_s, ln1_b, qn_b);
  ln_kernel<<<kN * kL, 256, 0, stream>>>(value, lnkv_s, lnkv_b, vn_b);

  const long long LD = (long long)kL * kD;  // 1,048,576

  // ---- per-head projections (shared 64x64 weights) ----
  launch_gemm(stream, qn_b, false, wq_b, qh_b, nullptr,
              kN * kL * kH, 64, 64, 64, 64, 64, 0, 0, 0, 0, 0, 0, 1, 1, GF_CBF16);
  launch_gemm(stream, vn_b, false, wk_b, kh_b, nullptr,
              kN * kL * kH, 64, 64, 64, 64, 64, 0, 0, 0, 0, 0, 0, 1, 1, GF_CBF16);
  // vh: batched per (n,h), stored transposed -> vhT[n][h][d][q]
  launch_gemm(stream, vn_b, false, wv_b, vhT_b, nullptr,
              kL, 64, 64, kD, 64, kL,
              LD, 64, 0, 0, (long long)kH * 64 * kL, (long long)64 * kL,
              kN, kH, GF_CBF16 | GF_TRANSC);

  // ---- energy = qh @ kh^T, straight into attn output region ----
  launch_gemm(stream, qh_b, false, kh_b, attn, nullptr,
              kL, kL, 64, kD, kD, kL,
              LD, 64, LD, 64, (long long)kL * kL * kH, (long long)kL * kL,
              kN, kH, 0);

  // ---- mask + scale + softmax (in place; attn is a final output) ----
  softmax_kernel<<<kN * kH * kL, 256, 0, stream>>>(attn, mask);

  // ---- o = attn @ vh (A = f32 attn, converted on load) ----
  launch_gemm(stream, attn, true, vhT_b, o_b, nullptr,
              kL, 64, kL, kL, kL, kD,
              (long long)kL * kL * kH, (long long)kL * kL,
              LD, (long long)64 * kL, LD, 64,
              kN, kH, GF_CBF16);

  // ---- y1 = o @ w_out^T + b_out ----
  launch_gemm_big(stream, o_b, wout_b, y1_b, b_out,
                  kN * kL, kD, kD, kD, kD, kD, GF_CBF16);

  // ---- GRU gate 1: x = query, y = y1 ----
  launch_gemm_big(stream, y1_b, g1W_b, ywc, nullptr,
                  kN * kL, 3 * kD, kD, kD, kD, 3 * kD, 0);
  launch_gemm_big(stream, qx_b, g1U_b, xuc, nullptr,
                  kN * kL, 2 * kD, kD, kD, kD, 2 * kD, 0);
  gru_pre_kernel<<<4096, 256, 0, stream>>>(ywc, xuc, query, g1_bg, rx_b, zbuf);
  launch_gemm_big(stream, rx_b, g1U_b + 2 * (size_t)kD * kD, rxu, nullptr,
                  kN * kL, kD, kD, kD, kD, kD, 0);
  gru_comb_kernel<<<4096, 256, 0, stream>>>(ywc, rxu, zbuf, query, hbuf, h_b);

  // ---- ln2 + fc + relu ----
  ln_kernel<<<kN * kL, 256, 0, stream>>>(hbuf, ln2_s, ln2_b, hn_b);
  launch_gemm_big(stream, hn_b, fcw_b, f_b, fc_b,
                  kN * kL, kD, kD, kD, kD, kD, GF_CBF16 | GF_RELU);

  // ---- GRU gate 2: x = h, y = f ----
  launch_gemm_big(stream, f_b, g2W_b, ywc, nullptr,
                  kN * kL, 3 * kD, kD, kD, kD, 3 * kD, 0);
  launch_gemm_big(stream, h_b, g2U_b, xuc, nullptr,
                  kN * kL, 2 * kD, kD, kD, kD, 2 * kD, 0);
  gru_pre_kernel<<<4096, 256, 0, stream>>>(ywc, xuc, hbuf, g2_bg, rx_b, zbuf);
  launch_gemm_big(stream, rx_b, g2U_b + 2 * (size_t)kD * kD, rxu, nullptr,
                  kN * kL, kD, kD, kD, kD, kD, 0);
  gru_comb_kernel<<<4096, 256, 0, stream>>>(ywc, rxu, zbuf, hbuf, out, nullptr);
}
